// GNN_dgl_26456998543861
// MI455X (gfx1250) — matrix operand order repaired and checked by reading, not compiled, not run on hardware
//
#include <hip/hip_runtime.h>
#include <hip/hip_bf16.h>
#include <math.h>

// ---------------------------------------------------------------------------
// GAT (2-layer, H=4, d=32) for MI455X / gfx1250, wave32.
// GEMMs: fragments pre-packed to the ISA 7.12.2 WMMA lane layout so the inner
// loop is  2x global_load_b128 (A) + 2x global_load_b128 (B) + v_wmma.
// Edge phase: f32 global atomics (feat + accumulator live in the 192MB L2).
// ---------------------------------------------------------------------------

#define GAT_N   100000
#define GAT_E   1600000
#define GAT_H   4
#define GAT_IN  128
#define GAT_HID 32           // per-head dim, both layers
#define GAT_HD  128          // H * HID
#define NEG_SLOPE 0.2f

typedef __attribute__((ext_vector_type(16))) __bf16 bf16x16;
typedef __attribute__((ext_vector_type(8)))  float  f32x8;

#define CDIV(a, b) (((a) + (b) - 1) / (b))

// ---------------------------------------------------------------- helpers ---
__device__ inline void atomicMaxF(float* addr, float value) {
    // IEEE trick: signed-int max for >=0, unsigned-int min for <0.
    // Correct with the -inf (0xFF800000) initial value.
    if (value >= 0.0f)
        atomicMax((int*)addr, __float_as_int(value));
    else
        atomicMin((unsigned int*)addr, __float_as_uint(value));
}

// ----------------------------------------------------- fragment packing -----
// Packed A layout: P[((tileM*KB + kb)*32 + lane)*16 + j], j = lane's element.
// 16-bit A 16x32 (ISA 7.12.2): lane = hi*16 + mlo holds row tileM*16+mlo,
//   K offset = (j<8) ? hi*8 + j : 16 + hi*8 + (j-8), within K-block kb.
__global__ void k_pack_a(const float* __restrict__ A, __bf16* __restrict__ P,
                         int M, int K) {
    int i = blockIdx.x * blockDim.x + threadIdx.x;
    if (i >= M * K) return;
    int j    = i & 15;
    int lane = (i >> 4) & 31;
    int rest = i >> 9;                 // = tileM*KB + kb
    int KB   = K >> 5;
    int kb   = rest % KB;
    int t    = rest / KB;
    int mlo  = lane & 15, hi = lane >> 4;
    int row  = t * 16 + mlo;
    int k    = kb * 32 + ((j < 8) ? (hi * 8 + j) : (16 + hi * 8 + (j - 8)));
    P[i] = (__bf16)A[(size_t)row * K + k];
}

// Packed B layout: P[((tileN*KB + kb)*32 + lane)*16 + j].
// 16-bit B 32x16: lane = hi*16 + mlo holds column tileN*16+mlo,
//   K offset = (j<8) ? hi*8 + j : 16 + hi*8 + (j-8), within K-block kb.
__global__ void k_pack_b(const float* __restrict__ B, __bf16* __restrict__ P,
                         int K, int Ncols) {
    int i = blockIdx.x * blockDim.x + threadIdx.x;
    if (i >= K * Ncols) return;
    int j    = i & 15;
    int lane = (i >> 4) & 31;
    int rest = i >> 9;                 // = tileN*KB + kb
    int KB   = K >> 5;
    int kb   = rest % KB;
    int tn   = rest / KB;
    int mlo  = lane & 15, hi = lane >> 4;
    int k    = kb * 32 + ((j < 8) ? (hi * 8 + j) : (16 + hi * 8 + (j - 8)));
    int col  = tn * 16 + mlo;
    P[i] = (__bf16)B[(size_t)k * Ncols + col];
}

// ------------------------------------------------------------- WMMA GEMM ----
// C[M,Ncols](f32) = A[M,K] * B[K,Ncols], both pre-packed bf16 fragments.
// One wave per 16x16 output tile; inner loop: 64B loads + one v_wmma.
__global__ void k_gemm_bf16(const __bf16* __restrict__ Apack,
                            const __bf16* __restrict__ Bpack,
                            float* __restrict__ C,
                            int M, int K, int Ncols) {
    const int lane  = threadIdx.x & 31;
    const int wave  = threadIdx.x >> 5;
    const int gw    = blockIdx.x * (blockDim.x >> 5) + wave;
    const int tilesN = Ncols >> 4;
    const int tileM  = gw / tilesN;
    const int tileN  = gw % tilesN;
    if (tileM * 16 >= M) return;            // wave-uniform exit
    const int KB = K >> 5;

    const bf16x16* Ap = (const bf16x16*)Apack + ((size_t)tileM * KB) * 32 + lane;
    const bf16x16* Bp = (const bf16x16*)Bpack + ((size_t)tileN * KB) * 32 + lane;

    f32x8 acc = {};
    for (int kb = 0; kb < KB; ++kb) {
        bf16x16 a = Ap[(size_t)kb * 32];
        bf16x16 b = Bp[(size_t)kb * 32];
        acc = __builtin_amdgcn_wmma_f32_16x16x32_bf16(
                  false, a, false, b, (short)0, acc, false, false);
    }

    // C/D layout: VGPR r holds M = tileM*16 + hi*8 + r, N = tileN*16 + mlo
    const int mlo  = lane & 15, hi = lane >> 4;
    const int bcol = tileN * 16 + mlo;
    const int crow = tileM * 16 + hi * 8;
#pragma unroll
    for (int r = 0; r < 8; ++r)
        C[(size_t)(crow + r) * Ncols + bcol] = acc[r];
}

// ------------------------------------------------------- attention logits ---
// one thread per (node, head): el/er = <feat[n,h,:], al/ar[h,:]>
__global__ void k_logits(const float* __restrict__ feat,
                         const float* __restrict__ al,
                         const float* __restrict__ ar,
                         float* __restrict__ el, float* __restrict__ er,
                         int NH) {
    int i = blockIdx.x * blockDim.x + threadIdx.x;
    if (i >= NH) return;
    int hh = i & (GAT_H - 1);
    const float* f = feat + (size_t)i * GAT_HID;
    const float* a = al + hh * GAT_HID;
    const float* r = ar + hh * GAT_HID;
    float se = 0.f, sr = 0.f;
#pragma unroll
    for (int j = 0; j < GAT_HID; ++j) {
        float v = f[j];
        se += v * a[j];
        sr += v * r[j];
    }
    el[i] = se;
    er[i] = sr;
}

// ------------------------------------------------------------------- init ---
__global__ void k_init_ms(float* __restrict__ m, float* __restrict__ s, int n) {
    int i = blockIdx.x * blockDim.x + threadIdx.x;
    if (i < n) { m[i] = -__builtin_inff(); s[i] = 0.0f; }
}

__global__ void k_zero(float* __restrict__ p, int n) {
    int i = blockIdx.x * blockDim.x + threadIdx.x;
    if (i < n) p[i] = 0.0f;
}

// ------------------------------------------------------------ edge phase ----
// pass 1: leaky-relu logit per (edge, head) + segment max via float atomicMax
__global__ void k_edge_logit_max(const int* __restrict__ src,
                                 const int* __restrict__ dst,
                                 const float* __restrict__ el,
                                 const float* __restrict__ er,
                                 float* __restrict__ eact,
                                 float* __restrict__ m, int EH) {
    int i = blockIdx.x * blockDim.x + threadIdx.x;
    if (i >= EH) return;
    int e  = i >> 2;
    int hh = i & 3;
    int sn = src[e], dn = dst[e];
    float z = el[sn * GAT_H + hh] + er[dn * GAT_H + hh];
    float a = (z > 0.f) ? z : NEG_SLOPE * z;
    eact[i] = a;
    atomicMaxF(&m[dn * GAT_H + hh], a);
}

// pass 2: ee = exp(e - m[dst]) (in place), segment sum via atomicAdd
__global__ void k_edge_expsum(const int* __restrict__ dst,
                              float* __restrict__ eact,
                              const float* __restrict__ m,
                              float* __restrict__ s, int EH) {
    int i = blockIdx.x * blockDim.x + threadIdx.x;
    if (i >= EH) return;
    int e  = i >> 2;
    int hh = i & 3;
    int dn = dst[e];
    float md = m[dn * GAT_H + hh];
    if (md == -__builtin_inff()) md = 0.0f;     // nodes with no in-edges
    float ee = __expf(eact[i] - md);
    eact[i] = ee;
    atomicAdd(&s[dn * GAT_H + hh], ee);
}

// pass 3: alpha = ee / max(s[dst], 1e-9)  (in place; one divide per edge-head)
__global__ void k_edge_alpha(const int* __restrict__ dst,
                             float* __restrict__ eact,
                             const float* __restrict__ s, int EH) {
    int i = blockIdx.x * blockDim.x + threadIdx.x;
    if (i >= EH) return;
    int e  = i >> 2;
    int hh = i & 3;
    int dn = dst[e];
    eact[i] = eact[i] / fmaxf(s[dn * GAT_H + hh], 1e-9f);
}

// pass 4: out[dst] += alpha * feat[src]; 32 lanes/edge, float4 gather + atomics
__global__ void k_edge_scatter(const int* __restrict__ src,
                               const int* __restrict__ dst,
                               const float* __restrict__ alpha,
                               const float* __restrict__ feat,
                               float* __restrict__ acc, int E) {
    int t = blockIdx.x * blockDim.x + threadIdx.x;
    int e = t >> 5;                 // 32 lanes per edge
    if (e >= E) return;
    int lane = t & 31;
    int j  = lane * 4;              // 4 consecutive dwords of the 128-wide row
    int hh = j >> 5;
    int sn = src[e], dn = dst[e];
    float a = alpha[e * GAT_H + hh];
    const float4 f = *(const float4*)(feat + (size_t)sn * GAT_HD + j);
    float* dp = acc + (size_t)dn * GAT_HD + j;
    atomicAdd(dp + 0, a * f.x);
    atomicAdd(dp + 1, a * f.y);
    atomicAdd(dp + 2, a * f.z);
    atomicAdd(dp + 3, a * f.w);
}

// --------------------------------------------------------------- finalize ---
// layer 0: h = mean_h tanh(acc + b)
__global__ void k_finalize0(const float* __restrict__ acc,
                            const float* __restrict__ b,
                            float* __restrict__ h, int ND) {
    int i = blockIdx.x * blockDim.x + threadIdx.x;
    if (i >= ND) return;
    int node = i >> 5, d = i & 31;
    float sum = 0.f;
#pragma unroll
    for (int hh = 0; hh < GAT_H; ++hh)
        sum += tanhf(acc[(size_t)node * GAT_HD + hh * GAT_HID + d] +
                     b[hh * GAT_HID + d]);
    h[i] = sum * 0.25f;
}

// layer 1: out = mean_h (acc + b)
__global__ void k_finalize1(const float* __restrict__ acc,
                            const float* __restrict__ b,
                            float* __restrict__ out, int ND) {
    int i = blockIdx.x * blockDim.x + threadIdx.x;
    if (i >= ND) return;
    int node = i >> 5, d = i & 31;
    float sum = 0.f;
#pragma unroll
    for (int hh = 0; hh < GAT_H; ++hh)
        sum += acc[(size_t)node * GAT_HD + hh * GAT_HID + d] +
               b[hh * GAT_HID + d];
    out[i] = sum * 0.25f;
}

// ============================================================ launcher ======
extern "C" void kernel_launch(void* const* d_in, const int* in_sizes, int n_in,
                              void* d_out, int out_size, void* d_ws, size_t ws_size,
                              hipStream_t stream) {
    (void)in_sizes; (void)n_in; (void)out_size; (void)ws_size;

    const float* x   = (const float*)d_in[0];
    const int*   src = (const int*)  d_in[1];
    const int*   dst = (const int*)  d_in[2];
    const float* W0  = (const float*)d_in[3];
    const float* al0 = (const float*)d_in[4];
    const float* ar0 = (const float*)d_in[5];
    const float* b0  = (const float*)d_in[6];
    const float* W1  = (const float*)d_in[7];
    const float* al1 = (const float*)d_in[8];
    const float* ar1 = (const float*)d_in[9];
    const float* b1  = (const float*)d_in[10];
    float* out = (float*)d_out;

    const int N = GAT_N, E = GAT_E;
    const int ND  = N * GAT_HID;   // 3.2M
    const int NHD = N * GAT_HD;    // 12.8M
    const int NH  = N * GAT_H;     // 400K
    const int EH  = E * GAT_H;     // 6.4M

    // ---- workspace carve-up (~154 MB total) ----
    char* ws = (char*)d_ws;
    size_t off = 0;
    auto take = [&](size_t bytes) -> char* {
        char* p = ws + off;
        off += (bytes + 255) & ~(size_t)255;
        return p;
    };
    float*  feat   = (float*)take((size_t)NHD * 4);           // 51.2 MB (both layers)
    float*  outacc = (float*)take((size_t)NHD * 4);           // 51.2 MB (both layers)
    char*   shared = take((size_t)EH * 4);                    // 25.6 MB, dual-use:
    __bf16* Apack0 = (__bf16*)shared;                         //   packed x (GEMM0 only)
    float*  eact   = (float*)shared;                          //   per-edge e/ee/alpha
    float*  el     = (float*)take((size_t)NH * 4);
    float*  er     = (float*)take((size_t)NH * 4);
    float*  mbuf   = (float*)take((size_t)NH * 4);
    float*  sbuf   = (float*)take((size_t)NH * 4);
    float*  hbuf   = (float*)take((size_t)ND * 4);            // 12.8 MB
    __bf16* Apack1 = (__bf16*)take((size_t)ND * 2);           // 6.4 MB packed h
    __bf16* Bpack0 = (__bf16*)take((size_t)GAT_IN * GAT_HD * 2);
    __bf16* Bpack1 = (__bf16*)take((size_t)GAT_HID * GAT_HD * 2);

    const int T = 256;
    // gemm grids: 8 waves/block, one 16x16 tile per wave
    const int gemmBlocks = CDIV((N / 16) * (GAT_HD / 16), 8); // 6250

    // ---- fragment packing (cast fused in) ----
    k_pack_a<<<CDIV(N * GAT_IN, T), T, 0, stream>>>(x, Apack0, N, GAT_IN);
    k_pack_b<<<CDIV(GAT_IN * GAT_HD, T), T, 0, stream>>>(W0, Bpack0, GAT_IN, GAT_HD);
    k_pack_b<<<CDIV(GAT_HID * GAT_HD, T), T, 0, stream>>>(W1, Bpack1, GAT_HID, GAT_HD);

    // ================= layer 0 =================
    k_gemm_bf16<<<gemmBlocks, T, 0, stream>>>(Apack0, Bpack0, feat, N, GAT_IN, GAT_HD);
    k_logits<<<CDIV(NH, T), T, 0, stream>>>(feat, al0, ar0, el, er, NH);
    k_init_ms<<<CDIV(NH, T), T, 0, stream>>>(mbuf, sbuf, NH);
    k_zero<<<CDIV(NHD, T), T, 0, stream>>>(outacc, NHD);
    k_edge_logit_max<<<CDIV(EH, T), T, 0, stream>>>(src, dst, el, er, eact, mbuf, EH);
    k_edge_expsum<<<CDIV(EH, T), T, 0, stream>>>(dst, eact, mbuf, sbuf, EH);
    k_edge_alpha<<<CDIV(EH, T), T, 0, stream>>>(dst, eact, sbuf, EH);
    k_edge_scatter<<<CDIV(E * 32, T), T, 0, stream>>>(src, dst, eact, feat, outacc, E);
    k_finalize0<<<CDIV(ND, T), T, 0, stream>>>(outacc, b0, hbuf, ND);

    // ================= layer 1 =================
    k_pack_a<<<CDIV(ND, T), T, 0, stream>>>(hbuf, Apack1, N, GAT_HID);
    k_gemm_bf16<<<gemmBlocks, T, 0, stream>>>(Apack1, Bpack1, feat, N, GAT_HID, GAT_HD);
    k_logits<<<CDIV(NH, T), T, 0, stream>>>(feat, al1, ar1, el, er, NH);
    k_init_ms<<<CDIV(NH, T), T, 0, stream>>>(mbuf, sbuf, NH);
    k_zero<<<CDIV(NHD, T), T, 0, stream>>>(outacc, NHD);
    k_edge_logit_max<<<CDIV(EH, T), T, 0, stream>>>(src, dst, el, er, eact, mbuf, EH);
    k_edge_expsum<<<CDIV(EH, T), T, 0, stream>>>(dst, eact, mbuf, sbuf, EH);
    k_edge_alpha<<<CDIV(EH, T), T, 0, stream>>>(dst, eact, sbuf, EH);
    k_edge_scatter<<<CDIV(E * 32, T), T, 0, stream>>>(src, dst, eact, feat, outacc, E);
    k_finalize1<<<CDIV(ND, T), T, 0, stream>>>(outacc, b1, out, ND);
}
